// GATModel_82317343195215
// MI455X (gfx1250) — matrix operand order repaired and checked by reading, not compiled, not run on hardware
//
#include <hip/hip_runtime.h>
#include <math.h>

// ---------------- problem constants (from reference) ----------------
#define NN      50000     // nodes
#define EE      800000    // edges (before self loops)
#define ETOT    850000    // edges + self loops
#define HIDD    128       // hidden dim
#define NHEAD   16
#define HDIM    8         // HIDD / NHEAD
#define NEG_SLOPE 0.2f

typedef __attribute__((ext_vector_type(16))) _Float16 v16h;
typedef __attribute__((ext_vector_type(8)))  _Float16 v8h;
typedef __attribute__((ext_vector_type(8)))  float    v8f;

// ---------------- helpers ----------------
__device__ inline void atomicMaxF(float* addr, float val) {
  unsigned int* ua = (unsigned int*)addr;
  unsigned int old = *ua;
  for (;;) {
    float cur = __uint_as_float(old);
    if (cur >= val) break;
    unsigned int assumed = old;
    old = atomicCAS(ua, assumed, __float_as_uint(val));
    if (old == assumed) break;
  }
}

__device__ inline void edge_nodes(const int* __restrict__ ei, int e, int& s, int& d) {
  if (e < EE) { s = ei[e]; d = ei[EE + e]; }
  else        { s = e - EE; d = e - EE; }         // self loop
}

__device__ inline float edge_logit(const float* __restrict__ al_s,
                                   const float* __restrict__ al_d,
                                   int s, int d, int h) {
  float v = al_s[s * NHEAD + h] + al_d[d * NHEAD + h];
  return v > 0.0f ? v : NEG_SLOPE * v;            // leaky_relu
}

// ---------------- kernels ----------------

// Convert W (f32 row-major [k][c]) to f16 COLUMN-major Wt[c][k] so each
// lane's B fragment is two contiguous 8-half runs.
__global__ void k_cvt_f16_t(const float* __restrict__ W, _Float16* __restrict__ Wt) {
  int i = blockIdx.x * blockDim.x + threadIdx.x;   // i = c*128 + k
  if (i >= HIDD * HIDD) return;
  int c = i >> 7, k = i & (HIDD - 1);
  Wt[i] = (_Float16)W[k * HIDD + c];
}

// xh = x @ W via v_wmma_f32_16x16x32_f16.
// Block = 256 threads = 8 waves. All waves share one 16x128 A tile staged in
// LDS as f16 (4 KB); wave w computes the 16x16 tile at cols [16w, 16w+16).
// K = 128 -> 4 fully-unrolled WMMA steps.
// Fragment layouts per CDNA5 ISA 7.12.2 (wave32):
//   A (16-bit, 16x32): lane m=lane&15, halves = K runs [kk+8*hi,+8) and
//                      [kk+8*hi+16,+8)            -> two ds_load_b128
//   B (16-bit, 32x16): lane col, halves = K runs [kk+8*hi,+8), [kk+8*hi+16,+8)
//                      contiguous in col-major Wt -> two global_load_b128
//   D (f32): lanes 0-15 rows r, lanes 16-31 rows 8+r, col = lane&15
__global__ void k_gemm_wmma(const float* __restrict__ x,
                            const _Float16* __restrict__ Wt,
                            float* __restrict__ xh) {
  __shared__ _Float16 sA[16 * HIDD];
  const int tid  = threadIdx.x;
  const int wave = tid >> 5;
  const int lane = tid & 31;
  const int row0 = blockIdx.x * 16;        // 50000 % 16 == 0: no tail guards

  // Stage A tile: 2048 floats, 8 per thread, coalesced; convert to f16.
  const float* xt = x + (size_t)row0 * HIDD;
#pragma unroll
  for (int t = 0; t < 8; ++t) {
    int i = t * 256 + tid;
    sA[i] = (_Float16)xt[i];
  }
  __syncthreads();

  const int m   = lane & 15;
  const int hi  = lane >> 4;
  const int col = wave * 16 + m;
  const _Float16* wcol = Wt + (size_t)col * HIDD;

  v8f acc = {};
#pragma unroll
  for (int kk = 0; kk < HIDD; kk += 32) {
    union { v16h v; v8h h[2]; } a, b;
    const _Float16* ap = &sA[m * HIDD + kk + hi * 8];
    a.h[0] = *(const v8h*)(ap);            // K = kk+8hi    .. +7
    a.h[1] = *(const v8h*)(ap + 16);       // K = kk+8hi+16 .. +23
    const _Float16* bp = wcol + kk + hi * 8;
    b.h[0] = *(const v8h*)(bp);            // K = kk+8hi    .. +7
    b.h[1] = *(const v8h*)(bp + 16);       // K = kk+8hi+16 .. +23
    acc = __builtin_amdgcn_wmma_f32_16x16x32_f16(
        /*neg_a=*/false, a.v, /*neg_b=*/false, b.v,
        /*c_mod=*/(short)0, acc, /*reuse_a=*/false, /*reuse_b=*/false);
  }

  float* outp = xh + (size_t)(row0 + hi * 8) * HIDD + col;
#pragma unroll
  for (int r = 0; r < 8; ++r)
    outp[(size_t)r * HIDD] = acc[r];
}

// Per-layer scratch init: acc=0 (N*128), m=-inf, denom=0 (N*16)
__global__ void k_init(float* __restrict__ acc, float* __restrict__ mx,
                       float* __restrict__ denom) {
  int i = blockIdx.x * blockDim.x + threadIdx.x;
  if (i < NN * HIDD) acc[i] = 0.0f;
  if (i < NN * NHEAD) { mx[i] = -INFINITY; denom[i] = 0.0f; }
}

// al_s[n,h] = xh[n,h,:]·a_src[h,:]; al_d likewise
__global__ void k_attn_logits(const float* __restrict__ xh,
                              const float* __restrict__ a_src,
                              const float* __restrict__ a_dst,
                              float* __restrict__ al_s, float* __restrict__ al_d) {
  int i = blockIdx.x * blockDim.x + threadIdx.x;   // i = n*16 + h
  if (i >= NN * NHEAD) return;
  int h = i & (NHEAD - 1);
  const float* v  = xh + (size_t)(i >> 4) * HIDD + h * HDIM;
  const float* as = a_src + h * HDIM;
  const float* ad = a_dst + h * HDIM;
  float s = 0.0f, d = 0.0f;
#pragma unroll
  for (int c = 0; c < HDIM; ++c) { float xv = v[c]; s += xv * as[c]; d += xv * ad[c]; }
  al_s[i] = s; al_d[i] = d;
}

// Pass 1: segment max of leaky_relu logits over dst
__global__ void k_edge_max(const int* __restrict__ ei,
                           const float* __restrict__ al_s, const float* __restrict__ al_d,
                           float* __restrict__ mx) {
  int t = blockIdx.x * blockDim.x + threadIdx.x;   // t = e*16 + h
  if (t >= ETOT * NHEAD) return;
  int e = t >> 4, h = t & (NHEAD - 1);
  int s, d; edge_nodes(ei, e, s, d);
  atomicMaxF(&mx[d * NHEAD + h], edge_logit(al_s, al_d, s, d, h));
}

// Pass 2: denom[d,h] += exp(e - m[d,h])
__global__ void k_edge_sum(const int* __restrict__ ei,
                           const float* __restrict__ al_s, const float* __restrict__ al_d,
                           const float* __restrict__ mx, float* __restrict__ denom) {
  int t = blockIdx.x * blockDim.x + threadIdx.x;
  if (t >= ETOT * NHEAD) return;
  int e = t >> 4, h = t & (NHEAD - 1);
  int s, d; edge_nodes(ei, e, s, d);
  float ex = __expf(edge_logit(al_s, al_d, s, d, h) - mx[d * NHEAD + h]);
  atomicAdd(&denom[d * NHEAD + h], ex);
}

// Pass 3: acc[d,h,:] += xh[s,h,:] * alpha(e,h)
__global__ void k_edge_agg(const int* __restrict__ ei,
                           const float* __restrict__ al_s, const float* __restrict__ al_d,
                           const float* __restrict__ mx, const float* __restrict__ denom,
                           const float* __restrict__ xh, float* __restrict__ acc) {
  int t = blockIdx.x * blockDim.x + threadIdx.x;
  if (t >= ETOT * NHEAD) return;
  int e = t >> 4, h = t & (NHEAD - 1);
  int s, d; edge_nodes(ei, e, s, d);
  float ex = __expf(edge_logit(al_s, al_d, s, d, h) - mx[d * NHEAD + h]);
  float alpha = ex / (denom[d * NHEAD + h] + 1e-16f);
  const float* src = xh  + (size_t)s * HIDD + h * HDIM;
  float*       dst = acc + (size_t)d * HIDD + h * HDIM;
#pragma unroll
  for (int c = 0; c < HDIM; ++c) atomicAdd(&dst[c], src[c] * alpha);
}

// result = acc + bias, optionally ELU
__global__ void k_finalize(const float* __restrict__ acc, const float* __restrict__ bias,
                           float* __restrict__ result, int do_elu) {
  int i = blockIdx.x * blockDim.x + threadIdx.x;
  if (i >= NN * HIDD) return;
  float v = acc[i] + bias[i & (HIDD - 1)];
  if (do_elu) v = v > 0.0f ? v : (__expf(v) - 1.0f);
  result[i] = v;
}

// ---------------- launch ----------------
extern "C" void kernel_launch(void* const* d_in, const int* in_sizes, int n_in,
                              void* d_out, int out_size, void* d_ws, size_t ws_size,
                              hipStream_t stream) {
  const float* x    = (const float*)d_in[0];
  const int*   ei   = (const int*)  d_in[1];
  const float* W1   = (const float*)d_in[2];
  const float* as1  = (const float*)d_in[3];
  const float* ad1  = (const float*)d_in[4];
  const float* b1   = (const float*)d_in[5];
  const float* W2   = (const float*)d_in[6];
  const float* as2  = (const float*)d_in[7];
  const float* ad2  = (const float*)d_in[8];
  const float* b2   = (const float*)d_in[9];
  float* out = (float*)d_out;

  // workspace carve-out (256B aligned)
  char* ws = (char*)d_ws;
  size_t off = 0;
  auto carve = [&](size_t bytes) -> void* {
    void* p = ws + off;
    off += (bytes + 255) & ~(size_t)255;
    return p;
  };
  _Float16* Wt    = (_Float16*)carve(sizeof(_Float16) * HIDD * HIDD);
  float*    xh    = (float*)carve(sizeof(float) * (size_t)NN * HIDD);
  float*    acc   = (float*)carve(sizeof(float) * (size_t)NN * HIDD);
  float*    h1    = (float*)carve(sizeof(float) * (size_t)NN * HIDD);
  float*    al_s  = (float*)carve(sizeof(float) * (size_t)NN * NHEAD);
  float*    al_d  = (float*)carve(sizeof(float) * (size_t)NN * NHEAD);
  float*    mx    = (float*)carve(sizeof(float) * (size_t)NN * NHEAD);
  float*    denom = (float*)carve(sizeof(float) * (size_t)NN * NHEAD);

  const int TB = 256;
  const int gInit = (NN * HIDD + TB - 1) / TB;
  const int gAttn = (NN * NHEAD + TB - 1) / TB;
  const int gEdge = (ETOT * NHEAD + TB - 1) / TB;
  const int gGemm = NN / 16;                 // exact
  const int gCvt  = (HIDD * HIDD + TB - 1) / TB;

  auto run_layer = [&](const float* xin, const float* W, const float* asrc,
                       const float* adst, const float* bias, float* result,
                       int do_elu) {
    k_cvt_f16_t  <<<gCvt,  TB, 0, stream>>>(W, Wt);
    k_gemm_wmma  <<<gGemm, TB, 0, stream>>>(xin, Wt, xh);
    k_init       <<<gInit, TB, 0, stream>>>(acc, mx, denom);
    k_attn_logits<<<gAttn, TB, 0, stream>>>(xh, asrc, adst, al_s, al_d);
    k_edge_max   <<<gEdge, TB, 0, stream>>>(ei, al_s, al_d, mx);
    k_edge_sum   <<<gEdge, TB, 0, stream>>>(ei, al_s, al_d, mx, denom);
    k_edge_agg   <<<gEdge, TB, 0, stream>>>(ei, al_s, al_d, mx, denom, xh, acc);
    k_finalize   <<<gInit, TB, 0, stream>>>(acc, bias, result, do_elu);
  };

  run_layer(x,  W1, as1, ad1, b1, h1,  1);  // layer 1 + ELU
  run_layer(h1, W2, as2, ad2, b2, out, 0);  // layer 2
}
